// ColorDeformConv2d_37237366456935
// MI455X (gfx1250) — matrix operand order repaired and checked by reading, not compiled, not run on hardware
//
#include <hip/hip_runtime.h>
#include <math.h>
#include <stdint.h>

typedef __attribute__((ext_vector_type(16))) _Float16 v16h;
typedef __attribute__((ext_vector_type(8)))  float    v8f;
typedef __attribute__((ext_vector_type(4)))  unsigned v4u;

#define BATCH 4
#define CH    64
#define HH    128
#define WW    128
#define HW    (HH * WW)            // 16384
#define NPIX  (BATCH * HW)         // 65536
#define KDIM  576                  // CH * 9 taps
#define HP    130                  // padded spatial extent
#define PROW  (HP * HP)            // 16900 padded pixels per batch
#define PPIX  (BATCH * PROW)       // 67600

union FragH { v4u q[2]; v16h h; };   // 16 halfs = 32 bytes
union H8    { _Float16 h[8]; v4u q; };

// Async copy of 32 bytes (2 x b128) from global to LDS; instruction offset
// applies to both the LDS destination and the global source address.
__device__ __forceinline__ void async_b128_x2(uint32_t lds_off, const void* gptr) {
    asm volatile("global_load_async_to_lds_b128 %0, %1, off\n\t"
                 "global_load_async_to_lds_b128 %0, %1, off offset:16"
                 :: "v"(lds_off), "v"(gptr) : "memory");
}
__device__ __forceinline__ void wait_async0() {
    asm volatile("s_wait_asynccnt 0x0" ::: "memory");
}

// ---------------------------------------------------------------------------
// Weight prep: f32 -> f16, permuted to K = tap*64 + c (tap-major) for the
// implicit-GEMM 3x3 convs; w_conv stays K = c*9 + n (its native layout).
// ---------------------------------------------------------------------------
__global__ void prep_weights_kernel(const float* __restrict__ w_p,
                                    const float* __restrict__ w_m,
                                    const float* __restrict__ w_c,
                                    const float* __restrict__ w_conv,
                                    _Float16* __restrict__ Wpm,   // [32][576]
                                    _Float16* __restrict__ Wc,    // [576][576]
                                    _Float16* __restrict__ Wcv)   // [64][576]
{
    int idx = blockIdx.x * blockDim.x + threadIdx.x;

    if (idx < 32 * KDIM) {                       // rows 0-17 w_p, 18-26 w_m, 27-31 zero
        int j = idx / KDIM, k = idx % KDIM;
        int tap = k >> 6, c = k & 63;
        float v = 0.f;
        if (j < 18)       v = w_p[(j * 64 + c) * 9 + tap];
        else if (j < 27)  v = w_m[((j - 18) * 64 + c) * 9 + tap];
        Wpm[idx] = (_Float16)v;
    }
    if (idx < KDIM * KDIM) {
        int j = idx / KDIM, k = idx % KDIM;
        int tap = k >> 6, c = k & 63;
        Wc[idx] = (_Float16)w_c[(j * 64 + c) * 9 + tap];
    }
    if (idx < 64 * KDIM) {
        Wcv[idx] = (_Float16)w_conv[idx];
    }
}

// ---------------------------------------------------------------------------
// 1x1 conv over concat(x, ref) producing HALO-PADDED channels-last f16:
// fusedP[b][y][x][c], y,x in [0,130), halo ring = 0, interior = conv result.
// ---------------------------------------------------------------------------
__global__ __launch_bounds__(256)
void conv_cd_kernel(const float* __restrict__ x, const float* __restrict__ ref,
                    const float* __restrict__ w_cd, const float* __restrict__ b_cd,
                    _Float16* __restrict__ fusedP)
{
    __shared__ float lds[4][128];
    const int tid  = threadIdx.x;
    const int base = blockIdx.x * 4;             // 4 padded pixels per block

    for (int idx = tid; idx < 4 * 128; idx += 256) {
        const int sub = idx >> 7, k = idx & 127;
        const int pixp = base + sub;
        const int b = pixp / PROW, rem = pixp % PROW;
        const int y = rem / HP, xc = rem % HP;
        float v = 0.f;
        if (y >= 1 && y <= 128 && xc >= 1 && xc <= 128) {
            const int hw = (y - 1) * WW + (xc - 1);
            v = (k < 64) ? x  [((size_t)(b * 64 + k)      << 14) + hw]
                         : ref[((size_t)(b * 64 + (k-64)) << 14) + hw];
        }
        lds[sub][k] = v;
    }
    __syncthreads();

    const int sub = tid >> 6, c = tid & 63;
    const int pixp = base + sub;
    const int rem = pixp % PROW;
    const int y = rem / HP, xc = rem % HP;
    const bool interior = (y >= 1 && y <= 128 && xc >= 1 && xc <= 128);

    float acc = b_cd[c];
    const float* w = w_cd + c * 128;
    #pragma unroll 8
    for (int k = 0; k < 128; ++k) acc += w[k] * lds[sub][k];
    fusedP[(size_t)pixp * CH + c] = interior ? (_Float16)acc : (_Float16)0.f;
}

// ---------------------------------------------------------------------------
// Implicit-GEMM 3x3 conv via WMMA. Wave tile: 32(M) x 64(N=pixels), K = 576.
// Fully unrolled K-loop -> all tap offsets are immediates; halo-padded input
// -> unconditional B loads. 8 accumulators (2 A-frags x 4 pixel sub-tiles).
// MODE 0: offset rows + sigmoid(m) -> offm[27][NPIX] (f32)
// MODE 1: tanh(c_off) -> Xoff[pix][j] (f16), j = c*9 + n
// ---------------------------------------------------------------------------
template <int MODE>
__global__ __launch_bounds__(128)
void conv3x3_wmma_kernel(const _Float16* __restrict__ fusedP,
                         const _Float16* __restrict__ Wmat,
                         const float* __restrict__ bias0,
                         const float* __restrict__ bias1,
                         float* __restrict__ offm,
                         _Float16* __restrict__ Xoff)
{
    const int lane = threadIdx.x & 31;
    const int wv   = threadIdx.x >> 5;
    const int hi   = lane >> 4;                  // K-half select
    const int ln16 = lane & 15;
    const int pixbase = (blockIdx.x * 4 + wv) * 64;
    const int mbase   = blockIdx.y * 32;

    v8f acc[2][4] = {};

    int pb[4];
    const _Float16* bptr[4];
    #pragma unroll
    for (int t = 0; t < 4; ++t) {
        pb[t] = pixbase + 16 * t + ln16;
        const int b = pb[t] >> 14, yy = (pb[t] >> 7) & 127, xx = pb[t] & 127;
        bptr[t] = fusedP + ((size_t)(b * PROW + (yy + 1) * HP + (xx + 1))) * CH + 8 * hi;
    }
    const _Float16* aptr[2];
    #pragma unroll
    for (int a = 0; a < 2; ++a) {
        aptr[a] = Wmat + (size_t)(mbase + a * 16 + ln16) * KDIM + 8 * hi;
        __builtin_prefetch(aptr[a], 0, 3);
    }

    #pragma unroll
    for (int kt = 0; kt < 18; ++kt) {
        const int kbase = kt * 32;
        const int tap   = kt >> 1;               // k0 = kt*32+8*hi -> tap = kt/2
        const int dy    = tap / 3 - 1;
        const int dx    = tap % 3 - 1;
        const int boff  = (dy * HP + dx) * CH + (kt & 1) * 32;  // halfs, immediate

        FragH fa[2];
        #pragma unroll
        for (int a = 0; a < 2; ++a) {
            fa[a].q[0] = *(const v4u*)(aptr[a] + kbase);
            fa[a].q[1] = *(const v4u*)(aptr[a] + kbase + 16);
        }
        #pragma unroll
        for (int t = 0; t < 4; ++t) {
            FragH fb;
            fb.q[0] = *(const v4u*)(bptr[t] + boff);
            fb.q[1] = *(const v4u*)(bptr[t] + boff + 16);
            #pragma unroll
            for (int a = 0; a < 2; ++a)
                acc[a][t] = __builtin_amdgcn_wmma_f32_16x16x32_f16(
                                false, fa[a].h, false, fb.h, (short)0, acc[a][t], false, false);
        }
    }

    if (MODE == 0) {
        #pragma unroll
        for (int a = 0; a < 2; ++a) {
            const int j0 = mbase + a * 16 + 8 * hi;
            #pragma unroll
            for (int t = 0; t < 4; ++t) {
                const int pix = pb[t];
                #pragma unroll
                for (int r = 0; r < 8; ++r) {
                    const int j = j0 + r;
                    const float v = acc[a][t][r];
                    if (j < 18) {
                        offm[(size_t)j * NPIX + pix] = v + bias0[j];
                    } else if (j < 27) {
                        const float s = v + bias1[j - 18];
                        offm[(size_t)j * NPIX + pix] = 1.0f / (1.0f + expf(-s));
                    }
                }
            }
        }
    } else {
        #pragma unroll
        for (int a = 0; a < 2; ++a) {
            const int j0 = mbase + a * 16 + 8 * hi;
            #pragma unroll
            for (int t = 0; t < 4; ++t) {
                H8 o;
                #pragma unroll
                for (int r = 0; r < 8; ++r)
                    o.h[r] = (_Float16)tanhf(acc[a][t][r] + bias0[j0 + r]);
                *(v4u*)(Xoff + (size_t)pb[t] * KDIM + j0) = o.q;
            }
        }
    }
}

// ---------------------------------------------------------------------------
// Deformable bilinear sampling, fused with c_off add and modulation:
// Xoff[pix][c*9+n] = (bilinear(c,n) + Xoff[pix][c*9+n]) * m[n]
// ---------------------------------------------------------------------------
__global__ __launch_bounds__(256)
void sample_kernel(const float* __restrict__ x,
                   const float* __restrict__ offm,
                   _Float16* __restrict__ Xoff)
{
    const int idx = blockIdx.x * blockDim.x + threadIdx.x;
    if (idx >= NPIX * 9) return;
    const int pix = idx / 9, n = idx % 9;
    const int b  = pix >> 14;
    const int yy = (pix >> 7) & 127;
    const int xx = pix & 127;

    const float prow = (float)(yy + 1 + (n / 3) - 1) + offm[(size_t)n * NPIX + pix];
    const float pcol = (float)(xx + 1 + (n % 3) - 1) + offm[(size_t)(9 + n) * NPIX + pix];
    const float mval = offm[(size_t)(18 + n) * NPIX + pix];

    const float flr = floorf(prow), flc = floorf(pcol);
    const float lt_r = fminf(fmaxf(flr,       0.f), 129.f);
    const float lt_c = fminf(fmaxf(flc,       0.f), 129.f);
    const float rb_r = fminf(fmaxf(flr + 1.f, 0.f), 129.f);
    const float rb_c = fminf(fmaxf(flc + 1.f, 0.f), 129.f);
    const float pr   = fminf(fmaxf(prow,      0.f), 129.f);
    const float pc   = fminf(fmaxf(pcol,      0.f), 129.f);

    const float g_lt = (1.f + (lt_r - pr)) * (1.f + (lt_c - pc));
    const float g_rb = (1.f - (rb_r - pr)) * (1.f - (rb_c - pc));
    const float g_lb = (1.f + (lt_r - pr)) * (1.f - (rb_c - pc));
    const float g_rt = (1.f - (rb_r - pr)) * (1.f + (lt_c - pc));

    const int ilt_r = (int)lt_r, ilt_c = (int)lt_c;
    const int irb_r = (int)rb_r, irb_c = (int)rb_c;
    const bool vrl = (ilt_r >= 1 && ilt_r <= 128), vrr = (irb_r >= 1 && irb_r <= 128);
    const bool vcl = (ilt_c >= 1 && ilt_c <= 128), vcr = (irb_c >= 1 && irb_c <= 128);
    const bool v_lt = vrl && vcl, v_rb = vrr && vcr, v_lb = vrl && vcr, v_rt = vrr && vcl;
    const int o_lt = (ilt_r - 1) * WW + (ilt_c - 1);
    const int o_rb = (irb_r - 1) * WW + (irb_c - 1);
    const int o_lb = (ilt_r - 1) * WW + (irb_c - 1);
    const int o_rt = (irb_r - 1) * WW + (ilt_c - 1);

    const float* xb = x + ((size_t)b * 64) * HW;
    _Float16* xo = Xoff + (size_t)pix * KDIM + n;
    #pragma unroll 4
    for (int c = 0; c < 64; ++c) {
        const float* plane = xb + (size_t)c * HW;
        float v = 0.f;
        if (v_lt) v += g_lt * plane[o_lt];
        if (v_rb) v += g_rb * plane[o_rb];
        if (v_lb) v += g_lb * plane[o_lb];
        if (v_rt) v += g_rt * plane[o_rt];
        const float prev = (float)xo[c * 9];
        xo[c * 9] = (_Float16)((v + prev) * mval);
    }
}

// ---------------------------------------------------------------------------
// Final conv == GEMM: out[64][NPIX] = Wconv[64][576] x Xoff[576][NPIX].
// Block = 4 waves sharing one 64-pixel strip; wave wv owns output rows
// wv*16..wv*16+15 and async-stages B sub-tile t=wv into a double-buffered
// LDS slot (global_load_async_to_lds_b128 / ASYNCcnt); all waves consume the
// staged B via ds_load_b128. Ordering per step: wait(asynccnt) -> barrier ->
// issue(kt+1) -> compute(kt); safe with 2 buffers.
// ---------------------------------------------------------------------------
__global__ __launch_bounds__(128)
void final_wmma_kernel(const _Float16* __restrict__ Xoff,
                       const _Float16* __restrict__ Wcv,
                       float* __restrict__ out)
{
    __shared__ _Float16 smem[2 * 2048];          // 2 slots x 4KB
    const int lane = threadIdx.x & 31;
    const int wv   = threadIdx.x >> 5;
    const int hi   = lane >> 4;
    const int ln16 = lane & 15;
    const int pixbase = blockIdx.x * 64;

    v8f acc[4] = {};

    // A: 16 output rows per wave
    const _Float16* aptr = Wcv + (size_t)(wv * 16 + ln16) * KDIM + 16 * hi;
    __builtin_prefetch(aptr, 0, 3);

    // This lane stages the B fragment that compute-lane `lane` of every wave
    // needs for sub-tile t = wv: pixel pixbase+16*wv+(lane%16), k-half hi.
    const int spix = pixbase + 16 * wv + ln16;
    const _Float16* sptr = Xoff + (size_t)spix * KDIM + 16 * hi;
    const uint32_t lbase = (uint32_t)(size_t)(&smem[0]);
    const uint32_t lidx  = (uint32_t)(wv * 32 + lane) * 32;   // bytes within slot

    async_b128_x2(lbase + lidx, sptr);           // stage kt = 0 into slot 0

    int pb[4];
    #pragma unroll
    for (int t = 0; t < 4; ++t) pb[t] = pixbase + 16 * t + ln16;

    #pragma unroll
    for (int kt = 0; kt < 18; ++kt) {
        wait_async0();                           // this wave's staged data landed
        __syncthreads();                         // everyone's data landed
        if (kt < 17)
            async_b128_x2(lbase + (uint32_t)(((kt + 1) & 1) << 12) + lidx,
                          sptr + (kt + 1) * 32);

        const _Float16* sb = smem + ((kt & 1) << 11);   // slot base (halfs)
        FragH fa;
        fa.q[0] = *(const v4u*)(aptr + kt * 32);
        fa.q[1] = *(const v4u*)(aptr + kt * 32 + 8);
        #pragma unroll
        for (int t = 0; t < 4; ++t) {
            FragH fb;
            const _Float16* p = sb + (t * 32 + lane) * 16;  // 32B per lane
            fb.q[0] = *(const v4u*)(p);
            fb.q[1] = *(const v4u*)(p + 8);
            acc[t] = __builtin_amdgcn_wmma_f32_16x16x32_f16(
                         false, fa.h, false, fb.h, (short)0, acc[t], false, false);
        }
    }

    const int j0 = wv * 16 + 8 * hi;
    #pragma unroll
    for (int t = 0; t < 4; ++t) {
        const int pix = pb[t];
        const int b = pix >> 14, rem = pix & (HW - 1);
        #pragma unroll
        for (int r = 0; r < 8; ++r) {
            const int oc = j0 + r;
            out[(((size_t)(b << 6) + oc) << 14) + rem] = acc[t][r];
        }
    }
}

// ---------------------------------------------------------------------------
extern "C" void kernel_launch(void* const* d_in, const int* in_sizes, int n_in,
                              void* d_out, int out_size, void* d_ws, size_t ws_size,
                              hipStream_t stream)
{
    const float* x      = (const float*)d_in[0];
    const float* ref    = (const float*)d_in[1];
    const float* w_cd   = (const float*)d_in[2];
    const float* b_cd   = (const float*)d_in[3];
    const float* w_p    = (const float*)d_in[4];
    const float* b_p    = (const float*)d_in[5];
    const float* w_m    = (const float*)d_in[6];
    const float* b_m    = (const float*)d_in[7];
    const float* w_c    = (const float*)d_in[8];
    const float* b_c    = (const float*)d_in[9];
    const float* w_conv = (const float*)d_in[10];
    float* out = (float*)d_out;

    char* ws = (char*)d_ws;
    _Float16* fusedP = (_Float16*)(ws);              //  8,652,800 B: [PPIX][64] f16 (halo)
    float*    offm   = (float*)   (ws +  8652800);   //  7,077,888 B: [27][NPIX] f32
    _Float16* Xoff   = (_Float16*)(ws + 15730688);   // 75,497,472 B: [NPIX][576] f16
    _Float16* Wpm    = (_Float16*)(ws + 91228160);   //     36,864 B: [32][576]
    _Float16* Wc     = (_Float16*)(ws + 91265024);   //    663,552 B: [576][576]
    _Float16* Wcv    = (_Float16*)(ws + 91928576);   //     73,728 B: [64][576]

    prep_weights_kernel<<<(KDIM * KDIM + 255) / 256, 256, 0, stream>>>(
        w_p, w_m, w_c, w_conv, Wpm, Wc, Wcv);

    conv_cd_kernel<<<PPIX / 4, 256, 0, stream>>>(x, ref, w_cd, b_cd, fusedP);

    dim3 gpm(256, 1);   // 32 output rows
    conv3x3_wmma_kernel<0><<<gpm, 128, 0, stream>>>(fusedP, Wpm, b_p, b_m, offm, nullptr);

    dim3 gc(256, 18);   // 576 output rows
    conv3x3_wmma_kernel<1><<<gc, 128, 0, stream>>>(fusedP, Wc, b_c, nullptr, nullptr, Xoff);

    sample_kernel<<<(NPIX * 9 + 255) / 256, 256, 0, stream>>>(x, offm, Xoff);

    dim3 gf(1024, 1);   // 1024 pixel strips, 64 output rows per block
    final_wmma_kernel<<<gf, 128, 0, stream>>>(Xoff, Wcv, out);
}
